// Mamba4KT_47364899340725
// MI455X (gfx1250) — matrix operand order repaired and checked by reading, not compiled, not run on hardware
//
#include <hip/hip_runtime.h>
#include <hip/hip_bf16.h>
#include <math.h>

#define BSZ     32
#define SEQL    512
#define TLEN    511
#define DMODEL  256
#define DINNER  512
#define DSTATE  16
#define DCONV   4
#define NHEAD   8
#define HDIM    32
#define NLAYER  2
#define NSKILL  500
#define MROWS   (BSZ * TLEN)   // 16352

typedef __attribute__((ext_vector_type(16))) __bf16 v16bf;
typedef __attribute__((ext_vector_type(2)))  __bf16 v2bf;
typedef __attribute__((ext_vector_type(8)))  float  v8f;

union BFV { v16bf v; __bf16 e[16]; v2bf p[8]; };

// Scalar fp32 -> bf16 (RNE) software fallback (matches LLVM's lowering).
__device__ __forceinline__ __bf16 f2bf(float f) {
    unsigned u = __builtin_bit_cast(unsigned, f);
    unsigned short s = (unsigned short)((u + 0x7fffu + ((u >> 16) & 1u)) >> 16);
    return __builtin_bit_cast(__bf16, s);
}

// Packed fp32x2 -> bf16x2 via the hardware packed convert. The gfx1250
// assembler validates the mnemonic; CDNA keeps V_CVT_PK_BF16_F32 and CDNA5
// retains the packed-bf16 VALU family (V_PK_*_BF16), so probe it directly.
__device__ __forceinline__ v2bf f2bf2(float a, float b) {
    unsigned r;
    asm("v_cvt_pk_bf16_f32 %0, %1, %2" : "=v"(r) : "v"(a), "v"(b));
    return __builtin_bit_cast(v2bf, r);
}

__device__ __forceinline__ v8f zero_v8f() {
    v8f z;
#pragma unroll
    for (int i = 0; i < 8; ++i) z[i] = 0.f;
    return z;
}

// ---------------------------------------------------------------------------
// WMMA fragment loaders (wave32, contiguous k = kbase+16g+e per lane half), fp32 -> bf16.
// ---------------------------------------------------------------------------

// A fragment 16x32 (MxK): lane holds row rbase+(lane&15); element e -> k =
// kbase+16g+e, g = lane>>4.
__device__ __forceinline__ v16bf load_A_frag(const float* __restrict__ A, int lda,
                                             int rbase, int kbase, int M, int K,
                                             float scale, int lane) {
    BFV u;
    int m = rbase + (lane & 15);
    int g = lane >> 4;
    if (m < M) {
        const float* row = A + (size_t)m * lda;
        int k0 = kbase + 16 * g;
        int k1 = kbase + 16 * g + 8;
        if (k0 + 8 <= K) {
            const float4* p = (const float4*)(row + k0);
            float4 a = p[0], b = p[1];
            u.p[0] = f2bf2(a.x * scale, a.y * scale);
            u.p[1] = f2bf2(a.z * scale, a.w * scale);
            u.p[2] = f2bf2(b.x * scale, b.y * scale);
            u.p[3] = f2bf2(b.z * scale, b.w * scale);
        } else {
#pragma unroll
            for (int e = 0; e < 8; ++e)
                u.e[e] = f2bf((k0 + e < K) ? row[k0 + e] * scale : 0.f);
        }
        if (k1 + 8 <= K) {
            const float4* p = (const float4*)(row + k1);
            float4 a = p[0], b = p[1];
            u.p[4] = f2bf2(a.x * scale, a.y * scale);
            u.p[5] = f2bf2(a.z * scale, a.w * scale);
            u.p[6] = f2bf2(b.x * scale, b.y * scale);
            u.p[7] = f2bf2(b.z * scale, b.w * scale);
        } else {
#pragma unroll
            for (int e = 0; e < 8; ++e)
                u.e[8 + e] = f2bf((k1 + e < K) ? row[k1 + e] * scale : 0.f);
        }
    } else {
#pragma unroll
        for (int q = 0; q < 8; ++q) u.p[q] = f2bf2(0.f, 0.f);
    }
    return u.v;
}

// B fragment 32x16 (KxN) from row-major W[N][ldw] (computes A @ W^T):
// lane holds column n = nbase+(lane&15); element e -> k = kbase + 16*g + e.
__device__ __forceinline__ v16bf load_B_frag(const float* __restrict__ W, int ldw,
                                             int nbase, int kbase, int N, int K,
                                             int lane) {
    BFV u;
    int n = nbase + (lane & 15);
    int k0 = kbase + ((lane >> 4) << 4);
    if (n < N) {
        const float* row = W + (size_t)n * ldw;
        if (k0 + 16 <= K) {
            const float4* p = (const float4*)(row + k0);
#pragma unroll
            for (int q = 0; q < 4; ++q) {
                float4 f = p[q];
                u.p[q * 2 + 0] = f2bf2(f.x, f.y);
                u.p[q * 2 + 1] = f2bf2(f.z, f.w);
            }
        } else {
#pragma unroll
            for (int e = 0; e < 16; ++e)
                u.e[e] = f2bf((k0 + e < K) ? row[k0 + e] : 0.f);
        }
    } else {
#pragma unroll
        for (int q = 0; q < 8; ++q) u.p[q] = f2bf2(0.f, 0.f);
    }
    return u.v;
}

// B fragment 32x16 from row-major V[T][ldv] at column block `col` (for P @ V):
// lane holds hd-column col+(lane&15); element e -> key row kbase + 16*g + e.
__device__ __forceinline__ v16bf load_V_frag(const float* __restrict__ V, int ldv,
                                             int kbase, int col, int lane) {
    BFV u;
    int n = col + (lane & 15);
    int k0 = kbase + ((lane >> 4) << 4);
#pragma unroll
    for (int q = 0; q < 8; ++q) {
        int ka = k0 + 2 * q;
        int kb = ka + 1;
        float va = (ka < TLEN) ? V[(size_t)ka * ldv + n] : 0.f;
        float vb = (kb < TLEN) ? V[(size_t)kb * ldv + n] : 0.f;
        u.p[q] = f2bf2(va, vb);
    }
    return u.v;
}

// ---------------------------------------------------------------------------
// Generic WMMA GEMM: C[M,N] = act(A[M,K] @ W[N,K]^T + bias)  (or C += A@W^T)
// block = 128 threads (4 waves); block tile 64M x 64N; each wave: 64M x 16N.
// ---------------------------------------------------------------------------
template <int ACT, bool ACCUM>
__global__ __launch_bounds__(128) void wmma_gemm_kernel(
        const float* __restrict__ A, int lda,
        const float* __restrict__ W, int ldw,
        const float* __restrict__ bias,
        float* __restrict__ C, int ldc,
        int M, int N, int K) {
    int lane = threadIdx.x & 31;
    int wave = threadIdx.x >> 5;
    int mblk = blockIdx.x * 64;
    int nblk = blockIdx.y * 64 + wave * 16;

    v8f acc0 = zero_v8f(), acc1 = zero_v8f(), acc2 = zero_v8f(), acc3 = zero_v8f();

    for (int k = 0; k < K; k += 32) {
        // Pull the next K-tile toward the WGP while this tile computes
        // (global_prefetch_b8; speculative, in-range clamped).
        if (k + 32 < K) {
            int pn = nblk + (lane & 15); if (pn >= N) pn = N - 1;
            __builtin_prefetch(W + (size_t)pn * ldw + k + 32, 0, 1);
            int pm = mblk + lane; if (pm >= M) pm = M - 1;
            __builtin_prefetch(A + (size_t)pm * lda + k + 32, 0, 1);
        }
        v16bf bf = load_B_frag(W, ldw, nblk, k, N, K, lane);
        v16bf a0 = load_A_frag(A, lda, mblk +  0, k, M, K, 1.f, lane);
        v16bf a1 = load_A_frag(A, lda, mblk + 16, k, M, K, 1.f, lane);
        v16bf a2 = load_A_frag(A, lda, mblk + 32, k, M, K, 1.f, lane);
        v16bf a3 = load_A_frag(A, lda, mblk + 48, k, M, K, 1.f, lane);
        acc0 = __builtin_amdgcn_wmma_f32_16x16x32_bf16(false, a0, false, bf, (short)0, acc0, false, false);
        acc1 = __builtin_amdgcn_wmma_f32_16x16x32_bf16(false, a1, false, bf, (short)0, acc1, false, false);
        acc2 = __builtin_amdgcn_wmma_f32_16x16x32_bf16(false, a2, false, bf, (short)0, acc2, false, false);
        acc3 = __builtin_amdgcn_wmma_f32_16x16x32_bf16(false, a3, false, bf, (short)0, acc3, false, false);
    }

    int g = lane >> 4;
    int col = nblk + (lane & 15);
    if (col >= N) return;
    float bv = (bias != nullptr) ? bias[col] : 0.f;

    auto store_tile = [&](const v8f& acc, int rbase) {
#pragma unroll
        for (int e = 0; e < 8; ++e) {
            int row = rbase + 8 * g + e;
            if (row < M) {
                size_t idx = (size_t)row * ldc + col;
                float v;
                if (ACCUM) {
                    v = C[idx] + acc[e];
                } else {
                    v = acc[e] + bv;
                    if (ACT == 1) v = tanhf(v);                                   // tanh
                    else if (ACT == 2) v = (v > 20.f) ? v : log1pf(__expf(v));    // softplus
                    else if (ACT == 3) v = 0.5f * v * (1.f + erff(v * 0.70710678118f)); // exact gelu
                }
                C[idx] = v;
            }
        }
    };
    store_tile(acc0, mblk);
    store_tile(acc1, mblk + 16);
    store_tile(acc2, mblk + 32);
    store_tile(acc3, mblk + 48);
}

// ---------------------------------------------------------------------------
// Embedding / gather kernel
// ---------------------------------------------------------------------------
__global__ void embed_kernel(const int* __restrict__ skills,
                             const int* __restrict__ responses,
                             const float* __restrict__ emb_concept,
                             const float* __restrict__ emb_answer,
                             const float* __restrict__ emb_true,
                             const float* __restrict__ emb_false,
                             float* __restrict__ xs, float* __restrict__ xst,
                             float* __restrict__ xsf) {
    int gid = blockIdx.x * blockDim.x + threadIdx.x;
    if (gid >= MROWS * DMODEL) return;
    int d = gid % DMODEL;
    int row = gid / DMODEL;
    int b = row / TLEN, t = row % TLEN;
    int c = skills[b * SEQL + t];
    int rr = responses[b * SEQL + t];
    int r = (rr > -1) ? rr : 0;   // masked_r = r * (r > -1)
    xs[gid]  = emb_answer[r * DMODEL + d] + emb_concept[c * DMODEL + d];
    xst[gid] = emb_true[(c * r + NSKILL * r) * DMODEL + d];
    xsf[gid] = emb_false[(c * (1 - r)) * DMODEL + d];
}

// ---------------------------------------------------------------------------
// Depthwise causal conv (kernel 4, left pad 3) + SiLU. xz is M x 1024; xc part
// is columns [0, 512).
// ---------------------------------------------------------------------------
__global__ void conv_silu_kernel(const float* __restrict__ xz,
                                 const float* __restrict__ cW,
                                 const float* __restrict__ cb,
                                 float* __restrict__ xc) {
    int gid = blockIdx.x * blockDim.x + threadIdx.x;
    if (gid >= MROWS * DINNER) return;
    int d = gid % DINNER;
    int row = gid / DINNER;
    int t = row % TLEN;
    float acc = cb[d];
#pragma unroll
    for (int j = 0; j < DCONV; ++j) {
        int tt = t - (DCONV - 1) + j;
        if (tt >= 0)
            acc += cW[d * DCONV + j] * xz[((size_t)row + (tt - t)) * (2 * DINNER) + d];
    }
    float sig = 1.f / (1.f + __expf(-acc));
    xc[(size_t)row * DINNER + d] = acc * sig;
}

// ---------------------------------------------------------------------------
// Selective scan. One block per batch element, one thread per channel d.
// B_t / C_t (16+16 floats) staged in LDS each step.  Also applies the skip
// (xc * Dv) and the SiLU(z) gate so out-proj can consume directly.
// ---------------------------------------------------------------------------
__global__ __launch_bounds__(512) void scan_kernel(
        const float* __restrict__ xc, const float* __restrict__ dtb,
        const float* __restrict__ xdbl, const float* __restrict__ xz,
        const float* __restrict__ A_log, const float* __restrict__ Dp,
        float* __restrict__ ymg) {
    __shared__ float BC[32];
    int b = blockIdx.x;
    int d = threadIdx.x;
    float Ac[DSTATE];
#pragma unroll
    for (int s = 0; s < DSTATE; ++s) Ac[s] = -__expf(A_log[d * DSTATE + s]);
    float Dv = Dp[d];
    float h[DSTATE];
#pragma unroll
    for (int s = 0; s < DSTATE; ++s) h[s] = 0.f;

    for (int t = 0; t < TLEN; ++t) {
        size_t row = (size_t)b * TLEN + t;
        if (d < 32) BC[d] = xdbl[row * 48 + 16 + d];
        __syncthreads();
        float dtt = dtb[row * DINNER + d];
        float xt  = xc[row * DINNER + d];
        float dx = dtt * xt;
        float y = 0.f;
#pragma unroll
        for (int s = 0; s < DSTATE; ++s) {
            h[s] = __expf(dtt * Ac[s]) * h[s] + dx * BC[s];
            y += h[s] * BC[16 + s];
        }
        float z = xz[row * (2 * DINNER) + DINNER + d];
        float sig = 1.f / (1.f + __expf(-z));
        ymg[row * DINNER + d] = (y + xt * Dv) * (z * sig);
        __syncthreads();
    }
}

// ---------------------------------------------------------------------------
// out = rmsnorm(a + b, w), row-wise over D=256. One block (256 thr) per row.
// ---------------------------------------------------------------------------
__global__ __launch_bounds__(256) void rmsnorm_add_kernel(
        const float* __restrict__ a, const float* __restrict__ bsrc,
        const float* __restrict__ w, float* __restrict__ out) {
    __shared__ float red[8];
    int row = blockIdx.x;
    int d = threadIdx.x;
    float x = a[(size_t)row * DMODEL + d] + bsrc[(size_t)row * DMODEL + d];
    float ss = x * x;
#pragma unroll
    for (int o = 16; o >= 1; o >>= 1) ss += __shfl_xor(ss, o);
    int wv = threadIdx.x >> 5;
    if ((threadIdx.x & 31) == 0) red[wv] = ss;
    __syncthreads();
    if (wv == 0) {
        float v = (threadIdx.x < 8) ? red[threadIdx.x] : 0.f;
#pragma unroll
        for (int o = 4; o >= 1; o >>= 1) v += __shfl_xor(v, o);
        if (threadIdx.x == 0) red[0] = v;
    }
    __syncthreads();
    float mean = red[0] / (float)DMODEL;
    out[(size_t)row * DMODEL + d] = x * rsqrtf(mean + 1e-12f) * w[d];
}

// ---------------------------------------------------------------------------
// Flash-style dual-value causal attention, WMMA for QK^T and PV.
// grid = (B*H, numQTiles), block = 1 wave (32 threads).
// Visits ALL key chunks with -1e9 masking so fully-masked rows (q==0) match
// the reference softmax-over-masked behavior exactly.
// ---------------------------------------------------------------------------
__global__ __launch_bounds__(32) void attention_kernel(
        const float* __restrict__ Y, const float* __restrict__ V1,
        const float* __restrict__ V2, float* __restrict__ O1,
        float* __restrict__ O2) {
    __shared__ float sP[16 * 33];   // 16x32 probs, padded stride to kill bank conflicts
    int lane = threadIdx.x;
    int b = blockIdx.x / NHEAD;
    int h = blockIdx.x % NHEAD;
    int qt = blockIdx.y;
    const float* Yb  = Y  + (size_t)b * TLEN * DMODEL;
    const float* V1b = V1 + (size_t)b * TLEN * DMODEL;
    const float* V2b = V2 + (size_t)b * TLEN * DMODEL;
    int colbase = h * HDIM;
    int g = lane >> 4;
    int cidx = lane & 15;

    const float QSCALE = 0.17677669529663689f;  // 1/sqrt(32)
    v16bf qf = load_A_frag(Yb + colbase, DMODEL, qt * 16, 0, TLEN, HDIM, QSCALE, lane);

    v8f o1a = zero_v8f(), o1b = zero_v8f(), o2a = zero_v8f(), o2b = zero_v8f();
    float mrow[8], lrow[8];
#pragma unroll
    for (int r = 0; r < 8; ++r) { mrow[r] = -3.0e38f; lrow[r] = 0.f; }

    for (int kb = 0; kb < TLEN; kb += 32) {
        v16bf kf0 = load_B_frag(Yb + colbase, DMODEL, kb,      0, TLEN, HDIM, lane);
        v16bf kf1 = load_B_frag(Yb + colbase, DMODEL, kb + 16, 0, TLEN, HDIM, lane);
        v8f s0 = __builtin_amdgcn_wmma_f32_16x16x32_bf16(false, qf, false, kf0, (short)0, zero_v8f(), false, false);
        v8f s1 = __builtin_amdgcn_wmma_f32_16x16x32_bf16(false, qf, false, kf1, (short)0, zero_v8f(), false, false);

#pragma unroll
        for (int r = 0; r < 8; ++r) {
            int qrow = qt * 16 + r + 8 * g;
            int key0 = kb + cidx;
            int key1 = kb + 16 + cidx;
            float a0 = s0[r], a1 = s1[r];
            if (key0 >= TLEN) a0 = -1e30f; else if (key0 >= qrow) a0 = -1e9f;
            if (key1 >= TLEN) a1 = -1e30f; else if (key1 >= qrow) a1 = -1e9f;
            float mx = fmaxf(a0, a1);
#pragma unroll
            for (int o = 8; o >= 1; o >>= 1) mx = fmaxf(mx, __shfl_xor(mx, o));
            float mn = fmaxf(mrow[r], mx);
            float p0 = __expf(a0 - mn);
            float p1 = __expf(a1 - mn);
            float rs = p0 + p1;
#pragma unroll
            for (int o = 8; o >= 1; o >>= 1) rs += __shfl_xor(rs, o);
            float alpha = __expf(mrow[r] - mn);
            lrow[r] = lrow[r] * alpha + rs;
            mrow[r] = mn;
            o1a[r] *= alpha; o1b[r] *= alpha; o2a[r] *= alpha; o2b[r] *= alpha;
            int prow = r + 8 * g;
            sP[prow * 33 + cidx]      = p0;
            sP[prow * 33 + 16 + cidx] = p1;
        }
        __syncthreads();
        // Re-read P as an A fragment (16x32), packed converts
        BFV pu;
#pragma unroll
        for (int q = 0; q < 4; ++q) {
            pu.p[q]     = f2bf2(sP[cidx * 33 + 16 * g + 2 * q],
                                sP[cidx * 33 + 16 * g + 2 * q + 1]);
            pu.p[q + 4] = f2bf2(sP[cidx * 33 + 16 * g + 8 + 2 * q],
                                sP[cidx * 33 + 16 * g + 8 + 2 * q + 1]);
        }
        v16bf vf;
        vf = load_V_frag(V1b, DMODEL, kb, colbase,      lane);
        o1a = __builtin_amdgcn_wmma_f32_16x16x32_bf16(false, pu.v, false, vf, (short)0, o1a, false, false);
        vf = load_V_frag(V1b, DMODEL, kb, colbase + 16, lane);
        o1b = __builtin_amdgcn_wmma_f32_16x16x32_bf16(false, pu.v, false, vf, (short)0, o1b, false, false);
        vf = load_V_frag(V2b, DMODEL, kb, colbase,      lane);
        o2a = __builtin_amdgcn_wmma_f32_16x16x32_bf16(false, pu.v, false, vf, (short)0, o2a, false, false);
        vf = load_V_frag(V2b, DMODEL, kb, colbase + 16, lane);
        o2b = __builtin_amdgcn_wmma_f32_16x16x32_bf16(false, pu.v, false, vf, (short)0, o2b, false, false);
        __syncthreads();
    }

#pragma unroll
    for (int r = 0; r < 8; ++r) {
        int t = qt * 16 + r + 8 * g;
        if (t < TLEN) {
            float inv = 1.f / lrow[r];
            size_t base = ((size_t)b * TLEN + t) * DMODEL + colbase;
            O1[base + cidx]      = o1a[r] * inv;
            O1[base + 16 + cidx] = o1b[r] * inv;
            O2[base + cidx]      = o2a[r] * inv;
            O2[base + 16 + cidx] = o2b[r] * inv;
        }
    }
}

// ---------------------------------------------------------------------------
// pred = sigmoid(dot(yc[row], out_W[cshft]) + out_b[cshft]); one wave per row.
// ---------------------------------------------------------------------------
__global__ __launch_bounds__(256) void pred_kernel(
        const float* __restrict__ yc, const int* __restrict__ skills,
        const float* __restrict__ outW, const float* __restrict__ outb,
        float* __restrict__ pred) {
    int wv = threadIdx.x >> 5;
    int lane = threadIdx.x & 31;
    int row = blockIdx.x * 8 + wv;
    if (row >= MROWS) return;
    int b = row / TLEN, t = row % TLEN;
    int cls = skills[b * SEQL + t + 1];   // cshft
    const float* yr = yc + (size_t)row * DMODEL;
    const float* wr = outW + (size_t)cls * DMODEL;
    float s = 0.f;
    for (int d = lane; d < DMODEL; d += 32) s += yr[d] * wr[d];
#pragma unroll
    for (int o = 16; o >= 1; o >>= 1) s += __shfl_xor(s, o);
    if (lane == 0) pred[row] = 1.f / (1.f + __expf(-(s + outb[cls])));
}

// ---------------------------------------------------------------------------
// Host-side GEMM dispatch
// ---------------------------------------------------------------------------
static void gemm_launch(hipStream_t s, const float* A, int lda, const float* W,
                        int ldw, const float* bias, float* C, int ldc, int M,
                        int N, int K, int act, bool accum) {
    dim3 grid((M + 63) / 64, (N + 63) / 64);
    dim3 block(128);
    if (accum) {
        wmma_gemm_kernel<0, true><<<grid, block, 0, s>>>(A, lda, W, ldw, bias, C, ldc, M, N, K);
        return;
    }
    switch (act) {
        case 1: wmma_gemm_kernel<1, false><<<grid, block, 0, s>>>(A, lda, W, ldw, bias, C, ldc, M, N, K); break;
        case 2: wmma_gemm_kernel<2, false><<<grid, block, 0, s>>>(A, lda, W, ldw, bias, C, ldc, M, N, K); break;
        case 3: wmma_gemm_kernel<3, false><<<grid, block, 0, s>>>(A, lda, W, ldw, bias, C, ldc, M, N, K); break;
        default: wmma_gemm_kernel<0, false><<<grid, block, 0, s>>>(A, lda, W, ldw, bias, C, ldc, M, N, K); break;
    }
}

extern "C" void kernel_launch(void* const* d_in, const int* in_sizes, int n_in,
                              void* d_out, int out_size, void* d_ws, size_t ws_size,
                              hipStream_t stream) {
    (void)in_sizes; (void)n_in; (void)out_size; (void)ws_size;

    const int*   skills      = (const int*)d_in[0];
    const int*   responses   = (const int*)d_in[2];
    const float* emb_concept = (const float*)d_in[3];
    const float* emb_answer  = (const float*)d_in[4];
    const float* emb_true    = (const float*)d_in[5];
    const float* emb_false   = (const float*)d_in[6];
    const float* mlp_W1      = (const float*)d_in[9];
    const float* mlp_b1      = (const float*)d_in[10];
    const float* mlp_W2      = (const float*)d_in[11];
    const float* mlp_b2      = (const float*)d_in[12];
    const float* in_W        = (const float*)d_in[13];
    const float* conv_W      = (const float*)d_in[14];
    const float* conv_b      = (const float*)d_in[15];
    const float* xproj_W     = (const float*)d_in[16];
    const float* dtproj_W    = (const float*)d_in[17];
    const float* dtproj_b    = (const float*)d_in[18];
    const float* A_log       = (const float*)d_in[19];
    const float* Dp          = (const float*)d_in[20];
    const float* outproj_W   = (const float*)d_in[21];
    const float* mnorm_w     = (const float*)d_in[22];
    const float* ffn_W1      = (const float*)d_in[23];
    const float* ffn_b1      = (const float*)d_in[24];
    const float* ffn_W2      = (const float*)d_in[25];
    const float* ffn_b2      = (const float*)d_in[26];
    const float* fnorm_w     = (const float*)d_in[27];
    const float* final_W     = (const float*)d_in[28];
    const float* final_b     = (const float*)d_in[29];
    const float* out_W       = (const float*)d_in[30];
    const float* out_b       = (const float*)d_in[31];

    float* w = (float*)d_ws;
    const size_t Mf = (size_t)MROWS;
    float* y    = w;                 // M x 256 (embedding state, then running activations)
    float* stv  = w + Mf * 256;      // M x 256 (true-state values)
    float* sfv  = w + Mf * 512;      // M x 256 (false-state values)
    float* tmp  = w + Mf * 768;      // M x 256 (MLP hidden)
    float* xz   = w + Mf * 1024;     // M x 1024 (in-proj out; reused as FFN hidden)
    float* xc   = w + Mf * 2048;     // M x 512 (conv+silu out)
    float* xdbl = w + Mf * 2560;     // M x 48
    float* dtb  = w + Mf * 2608;     // M x 512 (softplus dt)
    float* ymg  = w + Mf * 3120;     // M x 512 (scan out, gated)
    float* ymo  = w + Mf * 3632;     // M x 256 (out-proj)
    float* h1   = w + Mf * 3888;     // M x 256
    float* ffno = w + Mf * 4144;     // M x 256
    float* o1   = w + Mf * 4400;     // M x 256
    float* o2   = w + Mf * 4656;     // M x 256
    float* yc   = w + Mf * 4912;     // M x 256  (total 5168*M floats)

    // 1) embeddings + gathers
    {
        int total = MROWS * DMODEL;
        embed_kernel<<<(total + 255) / 256, 256, 0, stream>>>(
            skills, responses, emb_concept, emb_answer, emb_true, emb_false,
            y, stv, sfv);
    }
    // 2) shared tanh-MLP on state / st / sf (in-place via tmp)
    gemm_launch(stream, y,   DMODEL, mlp_W1, DMODEL, mlp_b1, tmp, DMODEL, MROWS, DMODEL, DMODEL, 1, false);
    gemm_launch(stream, tmp, DMODEL, mlp_W2, DMODEL, mlp_b2, y,   DMODEL, MROWS, DMODEL, DMODEL, 0, false);
    gemm_launch(stream, stv, DMODEL, mlp_W1, DMODEL, mlp_b1, tmp, DMODEL, MROWS, DMODEL, DMODEL, 1, false);
    gemm_launch(stream, tmp, DMODEL, mlp_W2, DMODEL, mlp_b2, stv, DMODEL, MROWS, DMODEL, DMODEL, 0, false);
    gemm_launch(stream, sfv, DMODEL, mlp_W1, DMODEL, mlp_b1, tmp, DMODEL, MROWS, DMODEL, DMODEL, 1, false);
    gemm_launch(stream, tmp, DMODEL, mlp_W2, DMODEL, mlp_b2, sfv, DMODEL, MROWS, DMODEL, DMODEL, 0, false);

    // 3) Mamba layers
    for (int i = 0; i < NLAYER; ++i) {
        const float* inWi = in_W     + (size_t)i * 1024 * 256;
        const float* cWi  = conv_W   + (size_t)i * 512 * 4;
        const float* cbi  = conv_b   + (size_t)i * 512;
        const float* xpWi = xproj_W  + (size_t)i * 48 * 512;
        const float* dpWi = dtproj_W + (size_t)i * 512 * 16;
        const float* dpbi = dtproj_b + (size_t)i * 512;
        const float* Ali  = A_log    + (size_t)i * 512 * 16;
        const float* Dpi  = Dp       + (size_t)i * 512;
        const float* opWi = outproj_W + (size_t)i * 256 * 512;
        const float* mnwi = mnorm_w  + (size_t)i * 256;
        const float* f1Wi = ffn_W1   + (size_t)i * 1024 * 256;
        const float* f1bi = ffn_b1   + (size_t)i * 1024;
        const float* f2Wi = ffn_W2   + (size_t)i * 256 * 1024;
        const float* f2bi = ffn_b2   + (size_t)i * 256;
        const float* fnwi = fnorm_w  + (size_t)i * 256;

        gemm_launch(stream, y, DMODEL, inWi, DMODEL, nullptr, xz, 1024, MROWS, 1024, DMODEL, 0, false);
        {
            int total = MROWS * DINNER;
            conv_silu_kernel<<<(total + 255) / 256, 256, 0, stream>>>(xz, cWi, cbi, xc);
        }
        gemm_launch(stream, xc, DINNER, xpWi, DINNER, nullptr, xdbl, 48, MROWS, 48, DINNER, 0, false);
        gemm_launch(stream, xdbl, 48, dpWi, 16, dpbi, dtb, DINNER, MROWS, DINNER, 16, 2, false);
        scan_kernel<<<BSZ, DINNER, 0, stream>>>(xc, dtb, xdbl, xz, Ali, Dpi, ymg);
        gemm_launch(stream, ymg, DINNER, opWi, DINNER, nullptr, ymo, DMODEL, MROWS, DMODEL, DINNER, 0, false);
        rmsnorm_add_kernel<<<MROWS, 256, 0, stream>>>(ymo, y, mnwi, h1);
        gemm_launch(stream, h1, DMODEL, f1Wi, DMODEL, f1bi, xz, 1024, MROWS, 1024, DMODEL, 3, false);
        gemm_launch(stream, xz, 1024, f2Wi, 1024, f2bi, ffno, DMODEL, MROWS, DMODEL, 1024, 0, false);
        rmsnorm_add_kernel<<<MROWS, 256, 0, stream>>>(ffno, h1, fnwi, y);
    }

    // 4) dual causal attention (q=k=y, v1=stv, v2=sfv)
    attention_kernel<<<dim3(BSZ * NHEAD, (TLEN + 15) / 16), 32, 0, stream>>>(
        y, stv, sfv, o1, o2);

    // 5) final head on concat([y,o1,o2]) via three K=256 passes over final_W slices
    gemm_launch(stream, y,  DMODEL, final_W,       768, final_b, yc, DMODEL, MROWS, DMODEL, DMODEL, 0, false);
    gemm_launch(stream, o1, DMODEL, final_W + 256, 768, nullptr, yc, DMODEL, MROWS, DMODEL, DMODEL, 0, true);
    gemm_launch(stream, o2, DMODEL, final_W + 512, 768, nullptr, yc, DMODEL, MROWS, DMODEL, DMODEL, 0, true);

    // 6) per-token class gather + sigmoid
    pred_kernel<<<(MROWS + 7) / 8, 256, 0, stream>>>(yc, skills, out_W, out_b, (float*)d_out);
}
